// T5Attention_65859028517329
// MI455X (gfx1250) — compile-verified, matrix operand order checked by reading, not run on hardware
//
#include <hip/hip_runtime.h>
#include <hip/hip_bf16.h>

// ---------------- problem constants ----------------
#define BB   2
#define SS   2048
#define NH   16
#define HD   64
#define DIMD 1024
#define MTOT (BB * SS)          // 4096

typedef __attribute__((ext_vector_type(16))) _Float16 v16h;
typedef __attribute__((ext_vector_type(8)))  float    v8f;

union Frag {                    // 16 halves per lane = one WMMA A/B operand
    v16h     h;
    unsigned u[8];
};

// k index of the first half held in VGPR v (pair 2v,2v+1) for this half-wave.
// A-matrix 16x32 f16 layout: lane<16 holds K {0..7,16..23}, lane>=16 holds {8..15,24..31}.
__device__ __forceinline__ int kbase(int v, int sub) {
    return 2 * v + (v >= 4 ? 8 : 0) + sub * 8;
}

// ---- CDNA5 async global->LDS DMA (ASYNCcnt-tracked), 16B per lane ----
__device__ __forceinline__ void async_ld_b128(_Float16* lds_dst,
                                              const _Float16* gsrc) {
    // generic pointers to LDS carry the LDS byte offset in addr[31:0]
    unsigned l = (unsigned)(size_t)lds_dst;
    asm volatile("global_load_async_to_lds_b128 %0, %1, off"
                 :: "v"(l), "v"(gsrc)
                 : "memory");
}
#define S_WAIT_ASYNC(N) asm volatile("s_wait_asynccnt " #N ::: "memory")

// ---------------- elementwise casts ----------------
__global__ void cast_f32_f16(const float* __restrict__ src,
                             _Float16* __restrict__ dst, int n) {
    int i = blockIdx.x * blockDim.x + threadIdx.x;
    int stride = gridDim.x * blockDim.x;
    for (; i < n; i += stride) dst[i] = (_Float16)src[i];
}

// src [K][N] f32 -> dst [N][K] f16
__global__ void transpose_cast(const float* __restrict__ src,
                               _Float16* __restrict__ dst, int K, int N) {
    int n = blockIdx.x * 16 + threadIdx.x;
    int k = blockIdx.y * 16 + threadIdx.y;
    if (n < N && k < K)
        dst[(size_t)n * K + k] = (_Float16)src[(size_t)k * N + n];
}

// ---------------- WMMA GEMM ----------------
// C[M,N] = A[M,K] * B[K,N], with B supplied transposed (Bt[N][K]).
// Block tile 128x128, 8 waves (4 M x 2 N), each wave 32x64 = 2x4 WMMA tiles.
// Global->LDS staging uses async DMA with double buffering so the copy of
// tile i+1 overlaps the 8-WMMA burst on tile i.
#define BTILE 128
#define BK 32
#define LDT 40                  // LDS row stride in halves (bank-conflict pad)

#define OUT_BHSD 0
#define OUT_BHDS 1
#define OUT_MNF32 2

__global__ __launch_bounds__(256) void wmma_gemm(
        const _Float16* __restrict__ A,   // [M][K]
        const _Float16* __restrict__ Bt,  // [N][K]
        void* __restrict__ out, int M, int N, int K, int mode) {
    __shared__ _Float16 As[2][BTILE * LDT];
    __shared__ _Float16 Bs[2][BTILE * LDT];

    const int tid  = threadIdx.x;
    const int lane = tid & 31;
    const int wid  = tid >> 5;
    const int wm   = (wid & 3) * 32;      // wave M offset in block
    const int wn   = (wid >> 2) * 64;     // wave N offset in block
    const int r    = lane & 15;
    const int sub  = lane >> 4;
    const int m0   = blockIdx.y * BTILE;
    const int n0   = blockIdx.x * BTILE;

    // this thread's two 16B staging chunks (of 512 per tile)
    const int row0 = tid >> 2,           seg0 = (tid & 3) * 8;
    const int row1 = (tid + 256) >> 2,   seg1 = ((tid + 256) & 3) * 8;
    const _Float16* gA0 = A  + (size_t)(m0 + row0) * K + seg0;
    const _Float16* gA1 = A  + (size_t)(m0 + row1) * K + seg1;
    const _Float16* gB0 = Bt + (size_t)(n0 + row0) * K + seg0;
    const _Float16* gB1 = Bt + (size_t)(n0 + row1) * K + seg1;
    const int lA0 = row0 * LDT + seg0, lA1 = row1 * LDT + seg1;

    v8f c[2][4];
#pragma unroll
    for (int tm = 0; tm < 2; ++tm)
#pragma unroll
        for (int tn = 0; tn < 4; ++tn)
#pragma unroll
            for (int rr = 0; rr < 8; ++rr) c[tm][tn][rr] = 0.0f;

    // prologue: DMA tile 0 into buffer 0
    async_ld_b128(&As[0][lA0], gA0);
    async_ld_b128(&As[0][lA1], gA1);
    async_ld_b128(&Bs[0][lA0], gB0);
    async_ld_b128(&Bs[0][lA1], gB1);

    int buf = 0;
    for (int kk = 0; kk < K; kk += BK) {
        // issue DMA for next tile into the other buffer, then wait for current
        if (kk + BK < K) {
            int nb = buf ^ 1;
            async_ld_b128(&As[nb][lA0], gA0 + kk + BK);
            async_ld_b128(&As[nb][lA1], gA1 + kk + BK);
            async_ld_b128(&Bs[nb][lA0], gB0 + kk + BK);
            async_ld_b128(&Bs[nb][lA1], gB1 + kk + BK);
            S_WAIT_ASYNC(4);               // current tile's 4 ops retired
        } else {
            S_WAIT_ASYNC(0);
        }
        __syncthreads();

        const _Float16* as = As[buf];
        const _Float16* bs = Bs[buf];
        Frag a[2], b[4];
#pragma unroll
        for (int tm = 0; tm < 2; ++tm)
#pragma unroll
            for (int v = 0; v < 8; ++v)
                a[tm].u[v] = *(const unsigned*)(as + (wm + tm * 16 + r) * LDT +
                                                kbase(v, sub));
#pragma unroll
        for (int tn = 0; tn < 4; ++tn)
#pragma unroll
            for (int v = 0; v < 8; ++v)
                b[tn].u[v] = *(const unsigned*)(bs + (wn + tn * 16 + r) * LDT +
                                                kbase(v, sub));
#pragma unroll
        for (int tm = 0; tm < 2; ++tm)
#pragma unroll
            for (int tn = 0; tn < 4; ++tn)
                c[tm][tn] = __builtin_amdgcn_wmma_f32_16x16x32_f16(
                    false, a[tm].h, false, b[tn].h, (short)0, c[tm][tn],
                    false, false);
        __syncthreads();   // all waves done reading before buf is overwritten
        buf ^= 1;
    }

    if (mode == OUT_MNF32) {
        float* o = (float*)out;
#pragma unroll
        for (int tm = 0; tm < 2; ++tm)
#pragma unroll
            for (int tn = 0; tn < 4; ++tn)
#pragma unroll
                for (int rr = 0; rr < 8; ++rr) {
                    int m = m0 + wm + tm * 16 + rr + 8 * sub;
                    int n = n0 + wn + tn * 16 + r;
                    o[(size_t)m * N + n] = c[tm][tn][rr];
                }
    } else if (mode == OUT_BHSD) {
        _Float16* o = (_Float16*)out;
#pragma unroll
        for (int tm = 0; tm < 2; ++tm)
#pragma unroll
            for (int tn = 0; tn < 4; ++tn)
#pragma unroll
                for (int rr = 0; rr < 8; ++rr) {
                    int m = m0 + wm + tm * 16 + rr + 8 * sub;
                    int n = n0 + wn + tn * 16 + r;
                    int bb = m / SS, s = m % SS, h = n / HD, d = n % HD;
                    o[(((size_t)(bb * NH + h)) * SS + s) * HD + d] =
                        (_Float16)c[tm][tn][rr];
                }
    } else {  // OUT_BHDS (V pre-transposed for the PV WMMA B-operand)
        _Float16* o = (_Float16*)out;
#pragma unroll
        for (int tm = 0; tm < 2; ++tm)
#pragma unroll
            for (int tn = 0; tn < 4; ++tn)
#pragma unroll
                for (int rr = 0; rr < 8; ++rr) {
                    int m = m0 + wm + tm * 16 + rr + 8 * sub;
                    int n = n0 + wn + tn * 16 + r;
                    int bb = m / SS, s = m % SS, h = n / HD, d = n % HD;
                    o[(((size_t)(bb * NH + h)) * HD + d) * SS + s] =
                        (_Float16)c[tm][tn][rr];
                }
    }
}

// ---------------- flash attention with T5 relative bias ----------------
#define QT 128

__global__ __launch_bounds__(256) void attn_kernel(
        const _Float16* __restrict__ Qh,   // [B,H,S,D]
        const _Float16* __restrict__ Kh,   // [B,H,S,D]
        const _Float16* __restrict__ Vh,   // [B,H,D,S]
        const float* __restrict__ relb,    // [32 buckets][NH]
        _Float16* __restrict__ ctx) {      // [B,S,NH*HD]
    __shared__ float    biasS[2 * SS - 1];     // bias vs (k - q) distance
    __shared__ _Float16 pb[8 * 16 * 32];       // per-wave P transpose buffer

    const int tid  = threadIdx.x;
    const int lane = tid & 31;
    const int wid  = tid >> 5;
    const int r    = lane & 15;
    const int sub  = lane >> 4;
    const int q0   = blockIdx.x * QT;
    const int bh   = blockIdx.y;
    const int b    = bh >> 4;
    const int h    = bh & 15;

    // T5 bucket bias table for this head (bidirectional, 32 buckets, maxdist 128)
    for (int i = tid; i < 2 * SS - 1; i += 256) {
        int d  = i - (SS - 1);               // k - q
        int rb = (d > 0) ? 16 : 0;
        int ad = d < 0 ? -d : d;
        int bucket;
        if (ad < 8) {
            bucket = rb + ad;
        } else {
            int lg = (int)(__logf((float)ad * 0.125f) *
                           (8.0f / 2.772588722f));  // /log(16) * 8
            int rl = 8 + lg;
            bucket = rb + (rl < 15 ? rl : 15);
        }
        biasS[i] = relb[bucket * NH + h];
    }
    __syncthreads();

    // Q fragments for this wave's 16 rows (two 32-wide k chunks of HD=64)
    Frag aQ[2];
    {
        size_t qb = (((size_t)bh) * SS + (q0 + wid * 16 + r)) * HD;
#pragma unroll
        for (int cchunk = 0; cchunk < 2; ++cchunk)
#pragma unroll
            for (int v = 0; v < 8; ++v)
                aQ[cchunk].u[v] =
                    *(const unsigned*)(Qh + qb + cchunk * 32 + kbase(v, sub));
    }

    float mrun[8], lrun[8];
    v8f   acc[4];
#pragma unroll
    for (int rr = 0; rr < 8; ++rr) { mrun[rr] = -1e30f; lrun[rr] = 0.0f; }
#pragma unroll
    for (int t = 0; t < 4; ++t)
#pragma unroll
        for (int rr = 0; rr < 8; ++rr) acc[t][rr] = 0.0f;

    _Float16* myp = pb + wid * 512;
    const int qrow_base = q0 + wid * 16 + 8 * sub;
    const size_t kfragbase = (((size_t)bh) * SS + r) * HD;
    const size_t vfragbase = (((size_t)bh) * HD + r) * SS;

    for (int s0 = 0; s0 < SS; s0 += 32) {
        // prefetch next chunk's K/V lines (global_prefetch_b8)
        if (s0 + 32 < SS) {
            __builtin_prefetch(Kh + kfragbase + (size_t)(s0 + 32) * HD, 0, 1);
            __builtin_prefetch(Vh + vfragbase + s0 + 32, 0, 1);
        }

        // ---- scores: Q (16x64) . K^T (64x32) as 2 n-tiles x 2 k-steps ----
        v8f sc[2];
#pragma unroll
        for (int t = 0; t < 2; ++t) {
            Frag bk0, bk1;
            size_t kb = kfragbase + (size_t)(s0 + t * 16) * HD;
#pragma unroll
            for (int v = 0; v < 8; ++v) {
                bk0.u[v] = *(const unsigned*)(Kh + kb + kbase(v, sub));
                bk1.u[v] = *(const unsigned*)(Kh + kb + 32 + kbase(v, sub));
            }
#pragma unroll
            for (int rr = 0; rr < 8; ++rr) sc[t][rr] = 0.0f;
            sc[t] = __builtin_amdgcn_wmma_f32_16x16x32_f16(
                false, aQ[0].h, false, bk0.h, (short)0, sc[t], false, false);
            sc[t] = __builtin_amdgcn_wmma_f32_16x16x32_f16(
                false, aQ[1].h, false, bk1.h, (short)0, sc[t], false, false);
        }

        // ---- add relative-position bias ----
#pragma unroll
        for (int t = 0; t < 2; ++t) {
            int kp = s0 + t * 16 + r;
#pragma unroll
            for (int rr = 0; rr < 8; ++rr)
                sc[t][rr] += biasS[kp - (qrow_base + rr) + (SS - 1)];
        }

        // ---- online softmax (row stats live in this lane's half-wave) ----
        float mnew[8], scale[8];
#pragma unroll
        for (int rr = 0; rr < 8; ++rr) {
            float x = fmaxf(sc[0][rr], sc[1][rr]);
#pragma unroll
            for (int off = 8; off; off >>= 1)
                x = fmaxf(x, __shfl_xor(x, off, 32));
            mnew[rr]  = fmaxf(mrun[rr], x);
            scale[rr] = __expf(mrun[rr] - mnew[rr]);
        }
#pragma unroll
        for (int rr = 0; rr < 8; ++rr) {
            sc[0][rr] = __expf(sc[0][rr] - mnew[rr]);
            sc[1][rr] = __expf(sc[1][rr] - mnew[rr]);
            float x = sc[0][rr] + sc[1][rr];
#pragma unroll
            for (int off = 8; off; off >>= 1) x += __shfl_xor(x, off, 32);
            lrun[rr] = lrun[rr] * scale[rr] + x;
            mrun[rr] = mnew[rr];
        }
#pragma unroll
        for (int t = 0; t < 4; ++t)
#pragma unroll
            for (int rr = 0; rr < 8; ++rr) acc[t][rr] *= scale[rr];

        // ---- transpose P (C-layout -> A-layout) through LDS ----
#pragma unroll
        for (int t = 0; t < 2; ++t)
#pragma unroll
            for (int rr = 0; rr < 8; ++rr)
                myp[(rr + 8 * sub) * 32 + t * 16 + r] = (_Float16)sc[t][rr];
        __syncthreads();

        Frag aP;
#pragma unroll
        for (int v = 0; v < 8; ++v)
            aP.u[v] = *(const unsigned*)(myp + r * 32 + kbase(v, sub));

        // ---- O += P (16x32) . V (32x64), V stored [B,H,D,S] ----
#pragma unroll
        for (int t = 0; t < 4; ++t) {
            Frag bv;
            size_t vb = vfragbase + (size_t)(t * 16) * SS + s0;
#pragma unroll
            for (int v = 0; v < 8; ++v)
                bv.u[v] = *(const unsigned*)(Vh + vb + kbase(v, sub));
            acc[t] = __builtin_amdgcn_wmma_f32_16x16x32_f16(
                false, aP.h, false, bv.h, (short)0, acc[t], false, false);
        }
        __syncthreads();
    }

    // ---- normalize and write context [B,S,NH*HD] f16 ----
#pragma unroll
    for (int rr = 0; rr < 8; ++rr) {
        float inv = 1.0f / lrun[rr];
        int   s   = qrow_base + rr;
#pragma unroll
        for (int t = 0; t < 4; ++t) {
            int d = t * 16 + r;
            ctx[((size_t)(b * SS + s)) * DIMD + h * HD + d] =
                (_Float16)(acc[t][rr] * inv);
        }
    }
}

// ---------------- host orchestration ----------------
extern "C" void kernel_launch(void* const* d_in, const int* in_sizes, int n_in,
                              void* d_out, int out_size, void* d_ws,
                              size_t ws_size, hipStream_t stream) {
    (void)in_sizes; (void)n_in; (void)out_size; (void)ws_size;
    const float* X    = (const float*)d_in[0];
    const float* Wq   = (const float*)d_in[1];
    const float* Wk   = (const float*)d_in[2];
    const float* Wv   = (const float*)d_in[3];
    const float* Wo   = (const float*)d_in[4];
    const float* relb = (const float*)d_in[5];

    char* ws = (char*)d_ws;
    const size_t MB = 1024 * 1024;
    _Float16* Xh  = (_Float16*)(ws);             //  8 MB: X f16 (reused as ctx)
    _Float16* WqT = (_Float16*)(ws +  8 * MB);   //  2 MB
    _Float16* WkT = (_Float16*)(ws + 10 * MB);   //  2 MB
    _Float16* WvT = (_Float16*)(ws + 12 * MB);   //  2 MB
    _Float16* WoT = (_Float16*)(ws + 14 * MB);   //  2 MB
    _Float16* Qh  = (_Float16*)(ws + 16 * MB);   //  8 MB [B,H,S,D]
    _Float16* Kh  = (_Float16*)(ws + 24 * MB);   //  8 MB [B,H,S,D]
    _Float16* Vh  = (_Float16*)(ws + 32 * MB);   //  8 MB [B,H,D,S]
    _Float16* Ctx = Xh;                          //  reuse after V GEMM

    cast_f32_f16<<<4096, 256, 0, stream>>>(X, Xh, MTOT * DIMD);

    dim3 tb(16, 16), tg(DIMD / 16, DIMD / 16);
    transpose_cast<<<tg, tb, 0, stream>>>(Wq, WqT, DIMD, DIMD);
    transpose_cast<<<tg, tb, 0, stream>>>(Wk, WkT, DIMD, DIMD);
    transpose_cast<<<tg, tb, 0, stream>>>(Wv, WvT, DIMD, DIMD);
    transpose_cast<<<tg, tb, 0, stream>>>(Wo, WoT, DIMD, DIMD);

    dim3 gg(DIMD / BTILE, MTOT / BTILE);         // (8, 32)
    wmma_gemm<<<gg, 256, 0, stream>>>(Xh, WqT, Qh, MTOT, DIMD, DIMD, OUT_BHSD);
    wmma_gemm<<<gg, 256, 0, stream>>>(Xh, WkT, Kh, MTOT, DIMD, DIMD, OUT_BHSD);
    wmma_gemm<<<gg, 256, 0, stream>>>(Xh, WvT, Vh, MTOT, DIMD, DIMD, OUT_BHDS);

    attn_kernel<<<dim3(SS / QT, BB * NH), 256, 0, stream>>>(Qh, Kh, Vh, relb,
                                                            Ctx);

    wmma_gemm<<<gg, 256, 0, stream>>>(Ctx, WoT, d_out, MTOT, DIMD, DIMD,
                                      OUT_MNF32);
}
